// MoEGPT_16174846837180
// MI455X (gfx1250) — matrix operand order repaired and checked by reading, not compiled, not run on hardware
//
#include <hip/hip_runtime.h>
#include <hip/hip_bf16.h>

// ---------------------------------------------------------------------------
// MoE-GPT forward for MI455X (gfx1250): bf16 WMMA GEMMs + fp32 accumulate.
// GEMM B operands pre-arranged as [N][K] so staging is pure 16B vector
// traffic; double-buffered LDS; async global->LDS DMA (ASYNCcnt path).
// ---------------------------------------------------------------------------

typedef unsigned short u16t;
typedef unsigned int   u32t;
typedef long long      ll_t;
typedef __attribute__((ext_vector_type(16))) __bf16 v16bf;
typedef __attribute__((ext_vector_type(8)))  float  v8f;
typedef __attribute__((ext_vector_type(4)))  int    v4i;

#define TPB 256

// CDNA5 async global->LDS DMA (ASYNCcnt) if the toolchain exposes it.
#ifndef USE_ASYNC_LDS
#if defined(__has_builtin)
#if __has_builtin(__builtin_amdgcn_global_load_async_to_lds_b128) && \
    __has_builtin(__builtin_amdgcn_s_wait_asynccnt)
#define USE_ASYNC_LDS 1
#endif
#endif
#endif
#ifndef USE_ASYNC_LDS
#define USE_ASYNC_LDS 0
#endif

#define AS1 __attribute__((address_space(1)))
#define AS3 __attribute__((address_space(3)))

__device__ __forceinline__ u16t f2bf(float f) {
  u32t u = __builtin_bit_cast(u32t, f);
  u32t r = u + 0x7FFFu + ((u >> 16) & 1u);   // round-to-nearest-even
  return (u16t)(r >> 16);
}

// ---------------------------------------------------------------------------
// GEMM: C[m,n] = alpha * sum_k A[m,k]*Bt[n,k] (+ bias[n])
// Requirements (all call sites satisfy): K % 32 == 0; rows 16B-aligned;
// M,N >= 1. Partial M/N tiles handled by row-clamping loads (duplicate rows
// feed garbage columns that the guarded epilogue never stores).
// Block tile 64x128, 8 waves, wave tile 32x32 = 2x2 WMMA(16x16x32 bf16).
// ---------------------------------------------------------------------------
__global__ __launch_bounds__(TPB)
void k_gemm(const u16t* __restrict__ A, const u16t* __restrict__ Bt,
            float* __restrict__ C, u16t* __restrict__ Cbf,
            const float* __restrict__ bias,
            int M, int N, int K, int lda, int ldb, int ldc, float alpha,
            int inner_count,
            ll_t a_os, ll_t a_is, ll_t b_os, ll_t b_is,
            ll_t c_os, ll_t c_is, ll_t bias_os) {
  const int z = blockIdx.z;
  const int outer = z / inner_count;
  const int inner = z % inner_count;
  A  += outer * a_os + inner * a_is;
  Bt += outer * b_os + inner * b_is;
  C  += outer * c_os + inner * c_is;
  if (Cbf)  Cbf  += outer * c_os + inner * c_is;
  if (bias) bias += outer * bias_os;

  __shared__ u16t As[2][64][40];    // [buf][m][k], padded rows (80B)
  __shared__ u16t Bs[2][128][40];   // [buf][n][k]

  const int m0 = blockIdx.y * 64;
  const int n0 = blockIdx.x * 128;
  const int tid  = threadIdx.x;
  const int lane = tid & 31;
  const int wave = tid >> 5;
  const int wm = (wave & 1) * 32;
  const int wn = (wave >> 1) * 32;

  // staging coords (fixed per thread); clamp rows instead of predicating
  const int arow = tid >> 2;             // 0..63
  const int akc  = (tid & 3) * 8;        // 0,8,16,24
  int gm = m0 + arow; if (gm >= M) gm = M - 1;
  const u16t* aptr = A + (ll_t)gm * lda;

  const int brow = tid >> 1;             // 0..127
  const int bkc  = (tid & 1) * 16;       // 0 or 16
  int gn = n0 + brow; if (gn >= N) gn = N - 1;
  const u16t* bptr = Bt + (ll_t)gn * ldb;

  v8f acc[2][2] = {};
  const int fm = lane & 15;
  const int hi = lane >> 4;              // 0: lanes 0-15, 1: lanes 16-31

#if USE_ASYNC_LDS
  auto stage = [&](int kk, int buf) {
    __builtin_amdgcn_global_load_async_to_lds_b128(
        (AS1 v4i*)(uintptr_t)(aptr + kk + akc),
        (AS3 v4i*)(u32t)(uintptr_t)&As[buf][arow][akc], 0, 0);
    __builtin_amdgcn_global_load_async_to_lds_b128(
        (AS1 v4i*)(uintptr_t)(bptr + kk + bkc),
        (AS3 v4i*)(u32t)(uintptr_t)&Bs[buf][brow][bkc], 0, 0);
    __builtin_amdgcn_global_load_async_to_lds_b128(
        (AS1 v4i*)(uintptr_t)(bptr + kk + bkc + 8),
        (AS3 v4i*)(u32t)(uintptr_t)&Bs[buf][brow][bkc + 8], 0, 0);
  };
  stage(0, 0);
  __builtin_amdgcn_s_wait_asynccnt(0);
  __syncthreads();
#else
  uint4 av  = *(const uint4*)(aptr + akc);
  uint4 bv0 = *(const uint4*)(bptr + bkc);
  uint4 bv1 = *(const uint4*)(bptr + bkc + 8);
  *(uint4*)&As[0][arow][akc]     = av;
  *(uint4*)&Bs[0][brow][bkc]     = bv0;
  *(uint4*)&Bs[0][brow][bkc + 8] = bv1;
  __syncthreads();
#endif

  for (int k0 = 0; k0 < K; k0 += 32) {
    const int  cur  = (k0 >> 5) & 1;
    const bool more = (k0 + 32) < K;

#if USE_ASYNC_LDS
    if (more) stage(k0 + 32, cur ^ 1);     // DMA next tile during compute
#else
    uint4 nav, nbv0, nbv1;
    if (more) {                            // issue next-tile loads early
      nav  = *(const uint4*)(aptr + k0 + 32 + akc);
      nbv0 = *(const uint4*)(bptr + k0 + 32 + bkc);
      nbv1 = *(const uint4*)(bptr + k0 + 32 + bkc + 8);
      if (k0 + 64 < K) {
        __builtin_prefetch(aptr + k0 + 64 + akc, 0, 1);  // global_prefetch_b8
        __builtin_prefetch(bptr + k0 + 64 + bkc, 0, 1);
      }
    }
#endif

    // ---- fragments per ISA 7.12.2 VGPR layouts (merge into ds_load_b128) -
    union Frag { u32t u[8]; v16bf v; };
    Frag a[2], b[2];
    const int abase = hi * 8;              // A: two 8-elem runs (abase, +16)
#pragma unroll
    for (int t = 0; t < 2; ++t) {
      const u16t* ap = &As[cur][wm + t * 16 + fm][0];
#pragma unroll
      for (int r = 0; r < 8; ++r) {
        int kk = abase + ((r < 4) ? (2 * r) : (16 + 2 * (r - 4)));
        a[t].u[r] = *(const u32t*)&ap[kk];
      }
    }
    const int bbase = hi * 8;              // B: two 8-elem runs (bbase, +16)
#pragma unroll
    for (int t = 0; t < 2; ++t) {
      const u16t* bp = &Bs[cur][wn + t * 16 + fm][bbase];
#pragma unroll
      for (int r = 0; r < 8; ++r) b[t].u[r] = *(const u32t*)&bp[(r < 4) ? (2 * r) : (16 + 2 * (r - 4))];
    }

#pragma unroll
    for (int i = 0; i < 2; ++i)
#pragma unroll
      for (int j = 0; j < 2; ++j)
        acc[i][j] = __builtin_amdgcn_wmma_f32_16x16x32_bf16(
            false, a[i].v, false, b[j].v, (short)0, acc[i][j], false, false);

#if USE_ASYNC_LDS
    if (more) __builtin_amdgcn_s_wait_asynccnt(0);
#else
    if (more) {
      *(uint4*)&As[cur ^ 1][arow][akc]     = nav;
      *(uint4*)&Bs[cur ^ 1][brow][bkc]     = nbv0;
      *(uint4*)&Bs[cur ^ 1][brow][bkc + 8] = nbv1;
    }
#endif
    __syncthreads();
  }

  // ---- store: C/D layout: lane n = fm, VGPR r -> m = r + hi*8 ------------
#pragma unroll
  for (int i = 0; i < 2; ++i) {
#pragma unroll
    for (int j = 0; j < 2; ++j) {
      int n = n0 + wn + j * 16 + fm;
      if (n >= N) continue;
      float bb = bias ? bias[n] : 0.0f;
#pragma unroll
      for (int r = 0; r < 8; ++r) {
        int m = m0 + wm + i * 16 + hi * 8 + r;
        if (m >= M) continue;
        float v = alpha * acc[i][j][r] + bb;
        C[(ll_t)m * ldc + n] = v;
        if (Cbf) Cbf[(ll_t)m * ldc + n] = f2bf(v);
      }
    }
  }
}

// ---------------------------------------------------------------------------
// Converts / transposes
// ---------------------------------------------------------------------------
__global__ void k_cvt(const float* __restrict__ s, u16t* __restrict__ d, ll_t n) {
  ll_t i = (ll_t)blockIdx.x * TPB + threadIdx.x;
  ll_t st = (ll_t)gridDim.x * TPB;
  for (; i < n; i += st) d[i] = f2bf(s[i]);
}

// Tiled transpose-convert: src fp32 [z][R][C] -> dst bf16 [z][C][R]
__global__ void k_cvt_t(const float* __restrict__ src, u16t* __restrict__ dst,
                        int R, int C, ll_t sb, ll_t db) {
  __shared__ float tile[32][33];
  src += (ll_t)blockIdx.z * sb;
  dst += (ll_t)blockIdx.z * db;
  int r0 = blockIdx.y * 32, c0 = blockIdx.x * 32;
  int tx = threadIdx.x & 31, ty = threadIdx.x >> 5;   // 256 thr: ty 0..7
  for (int i = ty; i < 32; i += 8) {
    int r = r0 + i, c = c0 + tx;
    tile[i][tx] = (r < R && c < C) ? src[(ll_t)r * C + c] : 0.f;
  }
  __syncthreads();
  for (int i = ty; i < 32; i += 8) {
    int c = c0 + i, r = r0 + tx;
    if (c < C && r < R) dst[(ll_t)c * R + r] = f2bf(tile[tx][i]);
  }
}

// Transpose V slice of qkv (fp32 [B*S][3H]) into vt bf16 [B][NH][DH][S]
__global__ void k_vt(const float* __restrict__ qkv, u16t* __restrict__ vt,
                     int S, int NH, int DH, int H3) {
  __shared__ float tile[32][33];
  int z = blockIdx.z;                    // b*NH + h
  int b = z / NH, h = z % NH;
  int s0 = blockIdx.x * 32, d0 = blockIdx.y * 32;
  int tx = threadIdx.x & 31, ty = threadIdx.x >> 5;
  const float* src = qkv + (ll_t)b * S * H3 + 2 * (NH * DH) + h * DH;
  for (int i = ty; i < 32; i += 8) {
    int s = s0 + i, d = d0 + tx;
    tile[i][tx] = (d < DH) ? src[(ll_t)s * H3 + d] : 0.f;
  }
  __syncthreads();
  u16t* dst = vt + (ll_t)z * DH * S;
  for (int i = ty; i < 32; i += 8) {
    int d = d0 + i, s = s0 + tx;
    if (d < DH) dst[(ll_t)d * S + s] = f2bf(tile[tx][i]);
  }
}

// ---------------------------------------------------------------------------
// Elementwise / reduction kernels
// ---------------------------------------------------------------------------
__global__ void k_embed(const int* __restrict__ ids, const float* __restrict__ tok,
                        const float* __restrict__ pos, float* __restrict__ x,
                        int S, int H) {
  int t = blockIdx.x;
  int s = t % S;
  int id = ids[t];
  for (int i = threadIdx.x; i < H; i += TPB)
    x[(ll_t)t * H + i] = tok[(ll_t)id * H + i] + pos[(ll_t)s * H + i];
}

__global__ void k_layernorm(const float* __restrict__ x, const float* __restrict__ sc,
                            const float* __restrict__ bi, float* __restrict__ of,
                            u16t* __restrict__ obf, int H) {
  int row = blockIdx.x;
  const float* xr = x + (ll_t)row * H;
  __shared__ float red[TPB];
  float s = 0.f;
  for (int i = threadIdx.x; i < H; i += TPB) s += xr[i];
  red[threadIdx.x] = s; __syncthreads();
  for (int o = TPB / 2; o > 0; o >>= 1) {
    if (threadIdx.x < o) red[threadIdx.x] += red[threadIdx.x + o];
    __syncthreads();
  }
  float mu = red[0] / H; __syncthreads();
  float v = 0.f;
  for (int i = threadIdx.x; i < H; i += TPB) { float d = xr[i] - mu; v += d * d; }
  red[threadIdx.x] = v; __syncthreads();
  for (int o = TPB / 2; o > 0; o >>= 1) {
    if (threadIdx.x < o) red[threadIdx.x] += red[threadIdx.x + o];
    __syncthreads();
  }
  float rstd = rsqrtf(red[0] / H + 1e-5f);
  for (int i = threadIdx.x; i < H; i += TPB) {
    float y = (xr[i] - mu) * rstd * sc[i] + bi[i];
    of[(ll_t)row * H + i] = y;
    if (obf) obf[(ll_t)row * H + i] = f2bf(y);
  }
}

__global__ void k_softmax(float* __restrict__ sc, u16t* __restrict__ pbf, int cols) {
  ll_t row = blockIdx.x;
  float* r = sc + row * cols;
  __shared__ float red[TPB];
  float m = -1e30f;
  for (int i = threadIdx.x; i < cols; i += TPB) m = fmaxf(m, r[i]);
  red[threadIdx.x] = m; __syncthreads();
  for (int o = TPB / 2; o > 0; o >>= 1) {
    if (threadIdx.x < o) red[threadIdx.x] = fmaxf(red[threadIdx.x], red[threadIdx.x + o]);
    __syncthreads();
  }
  m = red[0]; __syncthreads();
  float s = 0.f;
  for (int i = threadIdx.x; i < cols; i += TPB) { float e = __expf(r[i] - m); r[i] = e; s += e; }
  red[threadIdx.x] = s; __syncthreads();
  for (int o = TPB / 2; o > 0; o >>= 1) {
    if (threadIdx.x < o) red[threadIdx.x] += red[threadIdx.x + o];
    __syncthreads();
  }
  float inv = 1.0f / red[0];
  for (int i = threadIdx.x; i < cols; i += TPB) {
    float p = r[i] * inv;
    r[i] = p;
    pbf[row * cols + i] = f2bf(p);
  }
}

__global__ void k_add(float* __restrict__ x, const float* __restrict__ y, ll_t n) {
  ll_t i = (ll_t)blockIdx.x * TPB + threadIdx.x;
  ll_t st = (ll_t)gridDim.x * TPB;
  for (; i < n; i += st) x[i] += y[i];
}

__global__ void k_gelu(float* __restrict__ a, u16t* __restrict__ abf, ll_t n) {
  ll_t i = (ll_t)blockIdx.x * TPB + threadIdx.x;
  ll_t st = (ll_t)gridDim.x * TPB;
  for (; i < n; i += st) {
    float x = a[i];
    float y = 0.5f * x * (1.0f + erff(x * 0.70710678118654752f));  // exact gelu
    a[i] = y;
    abf[i] = f2bf(y);
  }
}

// one block per token: coalesced gate logits + top-1
__global__ void k_gate(const float* __restrict__ h, const float* __restrict__ gw,
                       float* __restrict__ gprob, int* __restrict__ idx, int H) {
  int t = blockIdx.x;
  const float* hr = h + (ll_t)t * H;
  float part[8];
#pragma unroll
  for (int e = 0; e < 8; ++e) part[e] = 0.f;
  for (int i = threadIdx.x; i < H; i += TPB) {
    float xv = hr[i];
#pragma unroll
    for (int e = 0; e < 8; ++e) part[e] += xv * gw[(ll_t)e * H + i];
  }
  __shared__ float red[TPB];
  __shared__ float lg[8];
  for (int e = 0; e < 8; ++e) {
    red[threadIdx.x] = part[e]; __syncthreads();
    for (int o = TPB / 2; o > 0; o >>= 1) {
      if (threadIdx.x < o) red[threadIdx.x] += red[threadIdx.x + o];
      __syncthreads();
    }
    if (threadIdx.x == 0) lg[e] = red[0];
    __syncthreads();
  }
  if (threadIdx.x == 0) {
    int am = 0; float mx = lg[0];
    for (int e = 1; e < 8; ++e) if (lg[e] > mx) { mx = lg[e]; am = e; }
    float se = 0.f;
    for (int e = 0; e < 8; ++e) se += __expf(lg[e] - mx);
    gprob[t] = 1.0f / se;               // softmax prob of argmax
    idx[t] = am;
  }
}

// Sequential top-1 capacity assignment (matches cumsum(mask)-1 semantics).
__global__ void k_dispatch(const int* __restrict__ idx, const float* __restrict__ gprob,
                           int* __restrict__ slot, int* __restrict__ loc,
                           float* __restrict__ cg, int T, int E, int CAP) {
  if (blockIdx.x != 0 || threadIdx.x != 0) return;
  int cnt[8];
  for (int e = 0; e < 8; ++e) cnt[e] = 0;
  for (int i = 0; i < E * CAP; ++i) slot[i] = -1;
  for (int t = 0; t < T; ++t) {
    int e = idx[t];
    int c = cnt[e]++;
    if (c < CAP) { slot[e * CAP + c] = t; loc[t] = c; cg[t] = gprob[t]; }
    else         { loc[t] = 0; cg[t] = 0.0f; }
  }
}

__global__ void k_gather(const int* __restrict__ slot, const float* __restrict__ h,
                         u16t* __restrict__ xsbf, int H) {
  int ec = blockIdx.x;
  int t = slot[ec];
  for (int i = threadIdx.x; i < H; i += TPB)
    xsbf[(ll_t)ec * H + i] = (t >= 0) ? f2bf(h[(ll_t)t * H + i]) : (u16t)0;
}

__global__ void k_combine(float* __restrict__ x, const float* __restrict__ h2,
                          const int* __restrict__ idx, const int* __restrict__ loc,
                          const float* __restrict__ cg, int H, int CAP) {
  int t = blockIdx.x;
  float g = cg[t];
  if (g == 0.0f) return;
  const float* src = h2 + ((ll_t)idx[t] * CAP + loc[t]) * H;
  for (int i = threadIdx.x; i < H; i += TPB) x[(ll_t)t * H + i] += g * src[i];
}

// ---------------------------------------------------------------------------
// Host orchestration
// ---------------------------------------------------------------------------
extern "C" void kernel_launch(void* const* d_in, const int* in_sizes, int n_in,
                              void* d_out, int out_size, void* d_ws, size_t ws_size,
                              hipStream_t stream) {
  constexpr int Vv = 32000, Ss = 1024, Hh = 768, NHh = 12, Ll = 2, Ee = 8, Bb = 2;
  constexpr int DHh = Hh / NHh;         // 64
  constexpr int FFf = 4 * Hh;           // 3072
  constexpr int Tt  = Bb * Ss;          // 2048
  constexpr int CAP = Tt / Ee;          // 256
  constexpr int H3  = 3 * Hh;           // 2304

  const int*   ids  = (const int*)  d_in[0];
  const float* temb = (const float*)d_in[1];
  const float* pemb = (const float*)d_in[2];
  const float* ln1s = (const float*)d_in[3];
  const float* ln1b = (const float*)d_in[4];
  const float* aiw  = (const float*)d_in[5];
  const float* aib  = (const float*)d_in[6];
  const float* aow  = (const float*)d_in[7];
  const float* aob  = (const float*)d_in[8];
  const float* ln2s = (const float*)d_in[9];
  const float* ln2b = (const float*)d_in[10];
  const float* gw   = (const float*)d_in[11];
  const float* w1   = (const float*)d_in[12];
  const float* b1   = (const float*)d_in[13];
  const float* w2   = (const float*)d_in[14];
  const float* b2   = (const float*)d_in[15];
  const float* lnfs = (const float*)d_in[16];
  const float* lnfb = (const float*)d_in[17];
  const float* hw   = (const float*)d_in[18];
  float* out = (float*)d_out;
  (void)in_sizes; (void)n_in; (void)out_size; (void)ws_size;

  // ---- workspace partitioning (256B aligned) -----------------------------
  char* wp = (char*)d_ws;
  auto alloc = [&](size_t bytes) -> void* {
    void* r = (void*)wp;
    wp += (bytes + 255) & ~(size_t)255;
    return r;
  };
  float* x      = (float*)alloc((size_t)Tt * Hh * 4);
  float* hf     = (float*)alloc((size_t)Tt * Hh * 4);
  u16t*  hbf    = (u16t*) alloc((size_t)Tt * Hh * 2);
  float* qkvf   = (float*)alloc((size_t)Tt * H3 * 4);
  u16t*  qkvbf  = (u16t*) alloc((size_t)Tt * H3 * 2);
  float* scores = (float*)alloc((size_t)Bb * NHh * Ss * Ss * 4);
  u16t*  probbf = (u16t*) alloc((size_t)Bb * NHh * Ss * Ss * 2);
  u16t*  vtbf   = (u16t*) alloc((size_t)Tt * Hh * 2);   // [B][NH][DH][S]
  float* ctxf   = (float*)alloc((size_t)Tt * Hh * 4);
  u16t*  ctxbf  = (u16t*) alloc((size_t)Tt * Hh * 2);
  float* aof    = (float*)alloc((size_t)Tt * Hh * 4);
  float* gprob  = (float*)alloc((size_t)Tt * 4);
  int*   eidx   = (int*)  alloc((size_t)Tt * 4);
  int*   eloc   = (int*)  alloc((size_t)Tt * 4);
  float* cg     = (float*)alloc((size_t)Tt * 4);
  int*   slot   = (int*)  alloc((size_t)Ee * CAP * 4);
  u16t*  xsbf   = (u16t*) alloc((size_t)Ee * CAP * Hh * 2);
  float* h1f    = (float*)alloc((size_t)Ee * CAP * FFf * 4);
  u16t*  h1bf   = (u16t*) alloc((size_t)Ee * CAP * FFf * 2);
  float* h2f    = (float*)alloc((size_t)Ee * CAP * Hh * 4);
  u16t*  aiwbf  = (u16t*) alloc((size_t)H3 * Hh * 2);
  u16t*  aowbf  = (u16t*) alloc((size_t)Hh * Hh * 2);
  u16t*  w1tbf  = (u16t*) alloc((size_t)Ee * FFf * Hh * 2);  // [E][FF][H]
  u16t*  w2tbf  = (u16t*) alloc((size_t)Ee * Hh * FFf * 2);  // [E][H][FF]
  u16t*  hwbf   = (u16t*) alloc((size_t)Vv * Hh * 2);

  auto cvt = [&](const float* s, u16t* d, ll_t n) {
    ll_t nb = (n + TPB - 1) / TPB; if (nb > 16384) nb = 16384;
    k_cvt<<<dim3((unsigned)nb), dim3(TPB), 0, stream>>>(s, d, n);
  };
  auto gemm = [&](const u16t* A, const u16t* B, float* C, u16t* Cbf,
                  const float* bias, int M, int N, int K, int lda, int ldb,
                  int ldc, float alpha, int batches, int inner,
                  ll_t aos, ll_t ais, ll_t bos, ll_t bis,
                  ll_t cos_, ll_t cis, ll_t bos2) {
    dim3 g((N + 127) / 128, (M + 63) / 64, batches);
    k_gemm<<<g, dim3(TPB), 0, stream>>>(A, B, C, Cbf, bias, M, N, K, lda,
        ldb, ldc, alpha, inner, aos, ais, bos, bis, cos_, cis, bos2);
  };

  // ---- embeddings + head weight conversion -------------------------------
  k_embed<<<dim3(Tt), dim3(TPB), 0, stream>>>(ids, temb, pemb, x, Ss, Hh);
  cvt(hw, hwbf, (ll_t)Vv * Hh);

  for (int l = 0; l < Ll; ++l) {
    // per-layer weight conversion (aiw/aow already [N][K]; w1/w2 transposed)
    cvt(aiw + (ll_t)l * H3 * Hh, aiwbf, (ll_t)H3 * Hh);
    cvt(aow + (ll_t)l * Hh * Hh, aowbf, (ll_t)Hh * Hh);
    k_cvt_t<<<dim3(FFf / 32, Hh / 32, Ee), dim3(TPB), 0, stream>>>(
        w1 + (ll_t)l * Ee * Hh * FFf, w1tbf, Hh, FFf,
        (ll_t)Hh * FFf, (ll_t)FFf * Hh);
    k_cvt_t<<<dim3(Hh / 32, FFf / 32, Ee), dim3(TPB), 0, stream>>>(
        w2 + (ll_t)l * Ee * FFf * Hh, w2tbf, FFf, Hh,
        (ll_t)FFf * Hh, (ll_t)Hh * FFf);

    // ---- attention block ----
    k_layernorm<<<dim3(Tt), dim3(TPB), 0, stream>>>(x, ln1s + (ll_t)l * Hh,
        ln1b + (ll_t)l * Hh, hf, hbf, Hh);
    // qkv = h @ in_w^T + in_b : [T,3H]
    gemm(hbf, aiwbf, qkvf, qkvbf, aib + (ll_t)l * H3,
         Tt, H3, Hh, Hh, Hh, H3, 1.0f, 1, 1, 0, 0, 0, 0, 0, 0, 0);
    // transpose V for probs@V
    k_vt<<<dim3(Ss / 32, DHh / 32, Bb * NHh), dim3(TPB), 0, stream>>>(
        qkvf, vtbf, Ss, NHh, DHh, H3);
    // scores = Q K^T / sqrt(DH), batched over (b, head)
    gemm(qkvbf, qkvbf + Hh, scores, nullptr, nullptr,
         Ss, Ss, DHh, H3, H3, Ss, 0.125f, Bb * NHh, NHh,
         (ll_t)Ss * H3, DHh, (ll_t)Ss * H3, DHh,
         (ll_t)NHh * Ss * Ss, (ll_t)Ss * Ss, 0);
    k_softmax<<<dim3(Bb * NHh * Ss), dim3(TPB), 0, stream>>>(scores, probbf, Ss);
    // ctx = probs @ V  (B = vt[z][DH][S], N=DH, K=S)
    gemm(probbf, vtbf, ctxf, ctxbf, nullptr,
         Ss, DHh, Ss, Ss, Ss, Hh, 1.0f, Bb * NHh, NHh,
         (ll_t)NHh * Ss * Ss, (ll_t)Ss * Ss,
         (ll_t)NHh * DHh * Ss, (ll_t)DHh * Ss,
         (ll_t)Ss * Hh, DHh, 0);
    // attn_out = ctx @ out_w^T + out_b ; residual
    gemm(ctxbf, aowbf, aof, nullptr, aob + (ll_t)l * Hh,
         Tt, Hh, Hh, Hh, Hh, Hh, 1.0f, 1, 1, 0, 0, 0, 0, 0, 0, 0);
    k_add<<<dim3(4096), dim3(TPB), 0, stream>>>(x, aof, (ll_t)Tt * Hh);

    // ---- MoE block ----
    k_layernorm<<<dim3(Tt), dim3(TPB), 0, stream>>>(x, ln2s + (ll_t)l * Hh,
        ln2b + (ll_t)l * Hh, hf, hbf, Hh);
    k_gate<<<dim3(Tt), dim3(TPB), 0, stream>>>(
        hf, gw + (ll_t)l * Ee * Hh, gprob, eidx, Hh);
    k_dispatch<<<dim3(1), dim3(1), 0, stream>>>(eidx, gprob, slot, eloc, cg,
        Tt, Ee, CAP);
    k_gather<<<dim3(Ee * CAP), dim3(TPB), 0, stream>>>(slot, hf, xsbf, Hh);
    // h1 = gelu(xs @ w1 + b1), per expert (B = w1t[e][FF][H])
    gemm(xsbf, w1tbf, h1f, nullptr, b1 + (ll_t)l * Ee * FFf,
         CAP, FFf, Hh, Hh, Hh, FFf, 1.0f, Ee, 1,
         (ll_t)CAP * Hh, 0, (ll_t)FFf * Hh, 0, (ll_t)CAP * FFf, 0, FFf);
    k_gelu<<<dim3(8192), dim3(TPB), 0, stream>>>(h1f, h1bf, (ll_t)Ee * CAP * FFf);
    // h2 = h1 @ w2 + b2, per expert (B = w2t[e][H][FF])
    gemm(h1bf, w2tbf, h2f, nullptr, b2 + (ll_t)l * Ee * Hh,
         CAP, Hh, FFf, FFf, FFf, Hh, 1.0f, Ee, 1,
         (ll_t)CAP * FFf, 0, (ll_t)Hh * FFf, 0, (ll_t)CAP * Hh, 0, Hh);
    k_combine<<<dim3(Tt), dim3(TPB), 0, stream>>>(x, h2f, eidx, eloc, cg, Hh, CAP);
  }

  // ---- final LN + vocab head --------------------------------------------
  k_layernorm<<<dim3(Tt), dim3(TPB), 0, stream>>>(x, lnfs, lnfb, hf, hbf, Hh);
  gemm(hbf, hwbf, out, nullptr, nullptr,
       Tt, Vv, Hh, Hh, Hh, Vv, 1.0f, 1, 1, 0, 0, 0, 0, 0, 0, 0);
}
